// DynamicAttention_54477365182625
// MI455X (gfx1250) — matrix operand order repaired and checked by reading, not compiled, not run on hardware
//
#include <hip/hip_runtime.h>
#include <math.h>

// ---------------------------------------------------------------------------
// DynamicAttention for MI455X (gfx1250, wave32, WMMA).
// f16 WMMA (f32 accumulate) for all GEMMs; fp32 for softmax/CDF math.
// ---------------------------------------------------------------------------

typedef _Float16 h16;
typedef __attribute__((ext_vector_type(8)))  _Float16 v8h;
typedef __attribute__((ext_vector_type(16))) _Float16 v16h;
typedef __attribute__((ext_vector_type(8)))  float    v8f;

#define SEQ 4096
#define DIM 512
#define INVNORM 0.04419417382415922f   // 1/sqrt(512)

// ---- WMMA wrapper ----------------------------------------------------------
__device__ __forceinline__ v8f wmma_f16(v16h a, v16h b, v8f c) {
  // v_wmma_f32_16x16x32_f16: D = A(16x32) * B(32x16) + C(16x16 f32)
  return __builtin_amdgcn_wmma_f32_16x16x32_f16(
      /*neg_a=*/false, a, /*neg_b=*/false, b,
      /*c_mod=*/(short)0, c, /*reuse_a=*/false, /*reuse_b=*/false);
}

// ---- Fragment loaders (ISA 7.12.2 layouts, wave32), vectorized -------------
// A 16x32 f16: lane L holds row L%16; lanes 0-15 get K {0..7,16..23}+k0,
// lanes 16-31 get K {8..15,24..31}+k0.  Two aligned 16B loads per lane.
__device__ __forceinline__ v16h load_a_frag(const h16* __restrict__ p,
                                            int m0, int ld, int k0) {
  const int lane = threadIdx.x & 31;
  const int hh = lane >> 4;
  const v8h* r = (const v8h*)(p + (size_t)(m0 + (lane & 15)) * ld + k0 + hh * 8);
  v8h lo = r[0];           // K = base+0..7
  v8h hi = r[2];           // K = base+16..23
  return __builtin_shufflevector(lo, hi, 0, 1, 2, 3, 4, 5, 6, 7,
                                 8, 9, 10, 11, 12, 13, 14, 15);
}

// B 32x16 f16 where column n of B is row (n0+n) of row-major M[*, ld]:
// lanes 0-15: K = k0+{0..7,16..23}, lanes 16-31: K = k0+{8..15,24..31}.  Two 16B loads.
__device__ __forceinline__ v16h load_bt_frag(const h16* __restrict__ p,
                                             int n0, int ld, int k0) {
  const int lane = threadIdx.x & 31;
  const int hh = lane >> 4;
  const v8h* r = (const v8h*)(p + (size_t)(n0 + (lane & 15)) * ld + k0 + hh * 8);
  v8h lo = r[0];
  v8h hi = r[2];
  return __builtin_shufflevector(lo, hi, 0, 1, 2, 3, 4, 5, 6, 7,
                                 8, 9, 10, 11, 12, 13, 14, 15);
}

// ---- Row-wise (16-lane half) wave primitives -------------------------------
__device__ __forceinline__ float rowmax16(float v) {
#pragma unroll
  for (int o = 1; o < 16; o <<= 1) v = fmaxf(v, __shfl_xor(v, o));
  return v;
}
__device__ __forceinline__ float rowsum16(float v) {
#pragma unroll
  for (int o = 1; o < 16; o <<= 1) v += __shfl_xor(v, o);
  return v;
}
__device__ __forceinline__ float rowscan16(float v) {  // inclusive prefix
  const int n = threadIdx.x & 15;
#pragma unroll
  for (int o = 1; o < 16; o <<= 1) {
    float t = __shfl_up(v, o);
    if (n >= o) v += t;
  }
  return v;
}

// ---- K0a: fp32 -> f16 convert (x) -----------------------------------------
__global__ __launch_bounds__(256) void cvt_kernel(const float* __restrict__ src,
                                                  h16* __restrict__ dst, int n) {
  int i = blockIdx.x * 256 + threadIdx.x;
  if (i < n) dst[i] = (h16)src[i];
}

// ---- K0b: fp32 -> f16 transposed convert (weights: [K,N] -> [N,K]) --------
__global__ __launch_bounds__(256) void cvt_w_kernel(const float* __restrict__ src,
                                                    h16* __restrict__ dstT) {
  int i = blockIdx.x * 256 + threadIdx.x;   // i < DIM*DIM
  int k = i >> 9, n = i & (DIM - 1);
  dstT[(size_t)n * DIM + k] = (h16)src[i];
}

// ---- K1: projections Y_p = X @ W_p (9 GEMMs); p==8 written transposed ------
__global__ __launch_bounds__(32) void proj_kernel(const h16* __restrict__ x16,
                                                  const h16* __restrict__ w16t,
                                                  h16* __restrict__ proj16,
                                                  h16* __restrict__ vt16) {
  const int p  = blockIdx.z;
  const int m0 = blockIdx.y * 16;
  const int n0 = blockIdx.x * 16;
  const h16* Wt = w16t + (size_t)p * DIM * DIM;   // [N, K] row-major
  v8f acc = {};
  for (int k0 = 0; k0 < DIM; k0 += 32)
    acc = wmma_f16(load_a_frag(x16, m0, DIM, k0),
                   load_bt_frag(Wt, n0, DIM, k0), acc);
  const int lane = threadIdx.x & 31;
  const int hh = lane >> 4;
  const int nn = n0 + (lane & 15);
  if (p == 8) {
    // V projection: store transposed [DIM, SEQ]; per lane the 8 accumulator
    // rows are contiguous in the transposed layout -> one packed 16B store.
    v8h hv;
#pragma unroll
    for (int r = 0; r < 8; ++r) hv[r] = (h16)acc[r];
    *(v8h*)(vt16 + (size_t)nn * SEQ + m0 + 8 * hh) = hv;
  } else {
    h16* Y = proj16 + (size_t)p * SEQ * DIM;
#pragma unroll
    for (int r = 0; r < 8; ++r)
      Y[(size_t)(m0 + r + 8 * hh) * DIM + nn] = (h16)acc[r];
  }
}

// ---- K2: pass A — per-row (max, sumexp) of masked left/right scores --------
__global__ __launch_bounds__(128) void passA_kernel(
    const h16* __restrict__ ql, const h16* __restrict__ kl,
    const h16* __restrict__ qr, const h16* __restrict__ kr,
    float* __restrict__ m_l, float* __restrict__ z_l,
    float* __restrict__ m_r, float* __restrict__ z_r) {
  const int bi = blockIdx.x;
  const int tid = threadIdx.x;
  const int w = tid >> 5;
  const int lane = tid & 31;
  const int hh = lane >> 4;
  const int n = lane & 15;
  const int m0 = bi * 16;
  __shared__ float sm[2][4][16], sz[2][4][16];

  float mm[2][8], zz[2][8];
#pragma unroll
  for (int s = 0; s < 2; ++s)
#pragma unroll
    for (int r = 0; r < 8; ++r) { mm[s][r] = -__builtin_inff(); zz[s][r] = 0.f; }

  for (int jt = w; jt < SEQ / 16; jt += 4) {
    const int colg = jt * 16 + n;
    if (jt <= bi) {  // left: support j <= i
      v8f acc = {};
      for (int k0 = 0; k0 < DIM; k0 += 32)
        acc = wmma_f16(load_a_frag(ql, m0, DIM, k0),
                       load_bt_frag(kl, jt * 16, DIM, k0), acc);
#pragma unroll
      for (int r = 0; r < 8; ++r) {
        const int rowg = m0 + r + 8 * hh;
        const bool valid = (colg <= rowg);
        float v = valid ? acc[r] * INVNORM : -__builtin_inff();
        float mnew = fmaxf(mm[0][r], rowmax16(v));
        float e = valid ? __expf(v - mnew) : 0.f;
        float s = rowsum16(e);
        zz[0][r] = zz[0][r] * __expf(mm[0][r] - mnew) + s;
        mm[0][r] = mnew;
      }
    }
    if (jt >= bi) {  // right: support j >= i
      v8f acc = {};
      for (int k0 = 0; k0 < DIM; k0 += 32)
        acc = wmma_f16(load_a_frag(qr, m0, DIM, k0),
                       load_bt_frag(kr, jt * 16, DIM, k0), acc);
#pragma unroll
      for (int r = 0; r < 8; ++r) {
        const int rowg = m0 + r + 8 * hh;
        const bool valid = (rowg <= colg);
        float v = valid ? acc[r] * INVNORM : -__builtin_inff();
        float mnew = fmaxf(mm[1][r], rowmax16(v));
        float e = valid ? __expf(v - mnew) : 0.f;
        float s = rowsum16(e);
        zz[1][r] = zz[1][r] * __expf(mm[1][r] - mnew) + s;
        mm[1][r] = mnew;
      }
    }
  }
  if (n == 0) {
#pragma unroll
    for (int r = 0; r < 8; ++r) {
      sm[0][w][r + 8 * hh] = mm[0][r]; sz[0][w][r + 8 * hh] = zz[0][r];
      sm[1][w][r + 8 * hh] = mm[1][r]; sz[1][w][r + 8 * hh] = zz[1][r];
    }
  }
  __syncthreads();
  if (tid < 32) {  // merge 4 wave partials (log-sum-exp merge)
    const int side = tid >> 4, row = tid & 15;
    float M = -__builtin_inff();
#pragma unroll
    for (int u = 0; u < 4; ++u) M = fmaxf(M, sm[side][u][row]);
    float Z = 0.f;
#pragma unroll
    for (int u = 0; u < 4; ++u) Z += sz[side][u][row] * __expf(sm[side][u][row] - M);
    if (side == 0) { m_l[m0 + row] = M; z_l[m0 + row] = Z; }
    else           { m_r[m0 + row] = M; z_r[m0 + row] = Z; }
  }
}

// ---- K3: pass B — left->right sweep: CDFs, score, online stats -------------
// 4 waves each produce one of {sl, sr, s_glb, s_loc} per j-tile into a
// double-buffered LDS tile; wave 0 runs the serial prefix-scan epilogue,
// overlapped with the next tile's WMMAs on waves 1-3.
__global__ __launch_bounds__(128) void passB_kernel(
    const h16* __restrict__ ql, const h16* __restrict__ kl,
    const h16* __restrict__ qr, const h16* __restrict__ kr,
    const h16* __restrict__ qg, const h16* __restrict__ kg,
    const h16* __restrict__ qc, const h16* __restrict__ kc,
    const float* __restrict__ m_l, const float* __restrict__ z_l,
    const float* __restrict__ m_r, const float* __restrict__ z_r,
    float* __restrict__ m_s, float* __restrict__ z_s,
    float* __restrict__ scores) {
  const int bi = blockIdx.x;
  const int tid = threadIdx.x;
  const int w = tid >> 5;
  const int lane = tid & 31;
  const int hh = lane >> 4;
  const int n = lane & 15;
  const int m0 = bi * 16;
  const int lanehi = (hh << 4) | 15;
  __shared__ float tiles[2][4][16][16];   // [buf][sl,sr,s_glb,s_loc][m][n]

  const h16* Aq = (w == 0) ? ql : (w == 1) ? qr : (w == 2) ? qg : qc;
  const h16* Bk = (w == 0) ? kl : (w == 1) ? kr : (w == 2) ? kg : kc;

  float mL[8], izL[8], mR[8], izR[8], pL[8], pR[8], mS[8], zS[8];
#pragma unroll
  for (int r = 0; r < 8; ++r) {
    const int rowg = m0 + r + 8 * hh;
    mL[r] = m_l[rowg]; izL[r] = 1.f / z_l[rowg];
    mR[r] = m_r[rowg]; izR[r] = 1.f / z_r[rowg];
    pL[r] = 0.f; pR[r] = 0.f;
    mS[r] = -__builtin_inff(); zS[r] = 0.f;
  }

  for (int jt = 0; jt < SEQ / 16; ++jt) {
    const int b = jt & 1;
    const bool active = (w == 0) ? (jt <= bi) : (w == 1) ? (jt >= bi) : true;
    v8f acc = {};
    if (active)
      for (int k0 = 0; k0 < DIM; k0 += 32)
        acc = wmma_f16(load_a_frag(Aq, m0, DIM, k0),
                       load_bt_frag(Bk, jt * 16, DIM, k0), acc);
#pragma unroll
    for (int r = 0; r < 8; ++r) tiles[b][w][r + 8 * hh][n] = acc[r];
    __syncthreads();

    if (w == 0) {  // serial epilogue: prefix CDFs + score + online softmax
      const int colg = jt * 16 + n;
#pragma unroll
      for (int r = 0; r < 8; ++r) {
        const int m = r + 8 * hh;
        const int rowg = m0 + m;
        // left CDF: prefix(exp)/Z
        float el = (colg <= rowg) ? __expf(tiles[b][0][m][n] * INVNORM - mL[r]) : 0.f;
        float cl = rowscan16(el);
        float lc = fminf((pL[r] + cl) * izL[r], 1.f);
        pL[r] += __shfl(cl, lanehi);
        // right CDF: 1 - exclusive_prefix(exp)/Z
        float er = (rowg <= colg) ? __expf(tiles[b][1][m][n] * INVNORM - mR[r]) : 0.f;
        float cr = rowscan16(er);
        float rc = fminf(fmaxf(1.f - (pR[r] + cr - er) * izR[r], 0.f), 1.f);
        pR[r] += __shfl(cr, lanehi);
        // combined score
        float sc = (tiles[b][2][m][n] + tiles[b][3][m][n] * (lc * rc)) * INVNORM;
        scores[(size_t)rowg * SEQ + colg] = sc;
        // online softmax stats for the final attention
        float mnew = fmaxf(mS[r], rowmax16(sc));
        float e = __expf(sc - mnew);
        float s = rowsum16(e);
        zS[r] = zS[r] * __expf(mS[r] - mnew) + s;
        mS[r] = mnew;
      }
    }
  }
  if (w == 0 && n == 0) {
#pragma unroll
    for (int r = 0; r < 8; ++r) {
      const int rowg = m0 + r + 8 * hh;
      m_s[rowg] = mS[r];
      z_s[rowg] = zS[r];
    }
  }
}

// ---- K4a: softmax normalize (in place) + f16 copy for output GEMM ----------
__global__ __launch_bounds__(256) void norm_kernel(const float* __restrict__ m_s,
                                                   const float* __restrict__ z_s,
                                                   float* __restrict__ attw,
                                                   h16* __restrict__ p16) {
  const int idx = blockIdx.x * 256 + threadIdx.x;
  const int i = idx >> 12;  // / SEQ
  const float v = __expf(attw[idx] - m_s[i]) * (1.f / z_s[i]);
  attw[idx] = v;
  p16[idx] = (h16)v;
}

// ---- K4b: att_output = att_weight @ V  (V held transposed [DIM,SEQ]) -------
__global__ __launch_bounds__(32) void outgemm_kernel(const h16* __restrict__ p16,
                                                     const h16* __restrict__ vt16,
                                                     float* __restrict__ out) {
  const int m0 = blockIdx.y * 16;
  const int n0 = blockIdx.x * 16;
  v8f acc = {};
  for (int k0 = 0; k0 < SEQ; k0 += 32)
    acc = wmma_f16(load_a_frag(p16, m0, SEQ, k0),
                   load_bt_frag(vt16, n0, SEQ, k0), acc);
  const int lane = threadIdx.x & 31;
  const int hh = lane >> 4;
  const int nn = n0 + (lane & 15);
#pragma unroll
  for (int r = 0; r < 8; ++r)
    out[(size_t)(m0 + r + 8 * hh) * DIM + nn] = acc[r];
}

// ---------------------------------------------------------------------------
extern "C" void kernel_launch(void* const* d_in, const int* in_sizes, int n_in,
                              void* d_out, int out_size, void* d_ws, size_t ws_size,
                              hipStream_t stream) {
  (void)in_sizes; (void)n_in; (void)out_size; (void)ws_size;
  const float* x = (const float*)d_in[0];
  float* att_out = (float*)d_out;                       // [SEQ, DIM]
  float* att_w   = att_out + (size_t)SEQ * DIM;         // [SEQ, SEQ] (also score scratch)

  char* ws = (char*)d_ws;
  h16* x16    = (h16*)ws; ws += (size_t)SEQ * DIM * sizeof(h16);
  h16* w16t   = (h16*)ws; ws += (size_t)9 * DIM * DIM * sizeof(h16);
  h16* proj16 = (h16*)ws; ws += (size_t)8 * SEQ * DIM * sizeof(h16);  // q/k mats
  h16* vt16   = (h16*)ws; ws += (size_t)DIM * SEQ * sizeof(h16);      // V^T
  h16* p16    = (h16*)ws; ws += (size_t)SEQ * SEQ * sizeof(h16);
  float* stats = (float*)ws;
  float* m_l = stats + 0 * SEQ; float* z_l = stats + 1 * SEQ;
  float* m_r = stats + 2 * SEQ; float* z_r = stats + 3 * SEQ;
  float* m_s = stats + 4 * SEQ; float* z_s = stats + 5 * SEQ;

  cvt_kernel<<<(SEQ * DIM + 255) / 256, 256, 0, stream>>>(x, x16, SEQ * DIM);
  for (int p = 0; p < 9; ++p)
    cvt_w_kernel<<<(DIM * DIM) / 256, 256, 0, stream>>>(
        (const float*)d_in[1 + p], w16t + (size_t)p * DIM * DIM);

  proj_kernel<<<dim3(DIM / 16, SEQ / 16, 9), 32, 0, stream>>>(x16, w16t, proj16, vt16);

  const h16* ql = proj16 + 0ull * SEQ * DIM;
  const h16* kl = proj16 + 1ull * SEQ * DIM;
  const h16* qr = proj16 + 2ull * SEQ * DIM;
  const h16* kr = proj16 + 3ull * SEQ * DIM;
  const h16* qg = proj16 + 4ull * SEQ * DIM;
  const h16* kg = proj16 + 5ull * SEQ * DIM;
  const h16* qc = proj16 + 6ull * SEQ * DIM;
  const h16* kc = proj16 + 7ull * SEQ * DIM;

  passA_kernel<<<SEQ / 16, 128, 0, stream>>>(ql, kl, qr, kr, m_l, z_l, m_r, z_r);
  passB_kernel<<<SEQ / 16, 128, 0, stream>>>(ql, kl, qr, kr, qg, kg, qc, kc,
                                             m_l, z_l, m_r, z_r, m_s, z_s, att_w);
  norm_kernel<<<(SEQ * SEQ) / 256, 256, 0, stream>>>(m_s, z_s, att_w, p16);
  outgemm_kernel<<<dim3(DIM / 16, SEQ / 16), 32, 0, stream>>>(p16, vt16, att_out);
}
